// LinOSSSequenceMixer_17626545783732
// MI455X (gfx1250) — matrix operand order repaired and checked by reading, not compiled, not run on hardware
//
#include <hip/hip_runtime.h>
#include <hip/hip_bf16.h>
#include <math.h>

// ---------------- problem constants ----------------
#define BATCH  16
#define SEQ    4096
#define HID    128
#define NSTATE 256
#define CHUNK  128
#define NCHUNK (SEQ / CHUNK)          // 32

typedef __attribute__((ext_vector_type(16))) _Float16 v16h;
typedef __attribute__((ext_vector_type(8)))  float    v8f;
typedef __attribute__((ext_vector_type(4)))  unsigned int u32x4;
typedef __attribute__((ext_vector_type(8)))  unsigned int u32x8;

// ---------------- workspace layout (floats) ----------------
#define P_M11   (0 * NSTATE)
#define P_M12   (1 * NSTATE)
#define P_M21   (2 * NSTATE)
#define P_M22   (3 * NSTATE)
#define P_C1    (4 * NSTATE)
#define P_C2    (5 * NSTATE)
#define P_MP11  (6 * NSTATE)
#define P_MP12  (7 * NSTATE)
#define P_MP21  (8 * NSTATE)
#define P_MP22  (9 * NSTATE)
#define CARRY_OFF 4096
#define CARRY_SZ  (BATCH * NCHUNK * NSTATE * 2)
#define BU_OFF    (CARRY_OFF + CARRY_SZ)

// ============================================================
// TDM: 2D tile Global -> LDS DMA (D# per CDNA5 ISA)
// tile_d0 x tile_d1 f32 tile, row stride stride_d0 elements,
// packed contiguously at lds_addr.
// ============================================================
__device__ __forceinline__ void tdm_load_2d_f32(unsigned int lds_addr,
                                                unsigned long long gaddr,
                                                unsigned int tensor_d0,
                                                unsigned int tensor_d1,
                                                unsigned int tile_d0,
                                                unsigned int tile_d1,
                                                unsigned int stride_d0) {
  u32x4 g0;
  g0.x = 1u;                                         // count=1 (valid), user mode
  g0.y = lds_addr;                                   // LDS byte address
  g0.z = (unsigned int)gaddr;                        // global_addr[31:0]
  g0.w = (unsigned int)((gaddr >> 32) & 0x1FFFFFFu)  // global_addr[56:32]
       | (2u << 30);                                 // type=2 ("image")
  u32x8 g1;
  g1.s0 = (2u << 16);                                // wg_mask=0, data_size=2 (4B)
  g1.s1 = (tensor_d0 & 0xFFFFu) << 16;               // tensor_dim0 lo @ bits 63:48
  g1.s2 = (tensor_d0 >> 16) | ((tensor_d1 & 0xFFFFu) << 16);
  g1.s3 = (tensor_d1 >> 16) | (tile_d0 << 16);       // dim1 hi | tile_dim0
  g1.s4 = tile_d1;                                   // tile_dim1 (tile_dim2=0 -> 2D)
  g1.s5 = stride_d0;                                 // tensor_dim0_stride[31:0]
  g1.s6 = 0u;
  g1.s7 = 0u;
  asm volatile("tensor_load_to_lds %0, %1" :: "s"(g0), "s"(g1) : "memory");
}

// ============================================================
// k0: per-state parameters + M^CHUNK by repeated squaring
// ============================================================
__global__ void k0_params(const float* __restrict__ A_diag,
                          const float* __restrict__ steps,
                          float* __restrict__ ws) {
  int n = threadIdx.x;
  float st    = 1.0f / (1.0f + __expf(-steps[n]));
  float A     = fmaxf(A_diag[n], 0.0f);
  float s2A   = st * st * A;
  float schur = 1.0f / (1.0f + s2A);
  float M11 = 1.0f - s2A * schur;
  float M12 = -st * A * schur;
  float M21 = st * schur;
  float M22 = schur;
  ws[P_M11 + n] = M11;  ws[P_M12 + n] = M12;
  ws[P_M21 + n] = M21;  ws[P_M22 + n] = M22;
  ws[P_C1  + n] = M11 * st;
  ws[P_C2  + n] = M21 * st;
  float a = M11, b = M12, c = M21, d = M22;
  #pragma unroll
  for (int i = 0; i < 7; ++i) {                      // 2^7 = CHUNK
    float na = a * a + b * c, nb = a * b + b * d;
    float nc = c * a + d * c, nd = c * b + d * d;
    a = na; b = nb; c = nc; d = nd;
  }
  ws[P_MP11 + n] = a;  ws[P_MP12 + n] = b;
  ws[P_MP21 + n] = c;  ws[P_MP22 + n] = d;
}

// ============================================================
// fragment builders (16-bit A 16x32, k = 16*(lane/16) + e, same as B),
// f32 LDS source, converted to f16 at read time
// ============================================================
__device__ __forceinline__ v16h build_afrag_f32(const float* tile /*[rows][32]*/,
                                                int lane, int row0) {
  v16h a;
  int M = (lane & 15) + row0;
  int kb_lo = (lane < 16) ? 0 : 16;
  int kb_hi = (lane < 16) ? 8 : 24;
  #pragma unroll
  for (int e = 0; e < 16; ++e) {
    int j = e >> 1, p = e & 1;
    int K = (j < 4) ? (kb_lo + 2 * j + p) : (kb_hi + 2 * (j - 4) + p);
    a[e] = (_Float16)tile[M * 32 + K];
  }
  return a;
}
__device__ __forceinline__ v16h build_bfrag_f32(const float* tile /*[cols][32]*/,
                                                int lane, int col0) {
  v16h b;
  int N  = lane & 15;
  int Kb = (lane < 16) ? 0 : 16;
  #pragma unroll
  for (int e = 0; e < 16; ++e)
    b[e] = (_Float16)tile[(col0 + N) * 32 + Kb + e];
  return b;
}

// ============================================================
// k1: Bu[b,t,n] = sum_h x[b,t,h]*B[n,h]
// 32 time rows/block, 256 thr (8 waves); wave -> 2 state x 2 time
// tiles; TDM double-buffered staging overlapped with WMMA.
// ============================================================
__global__ void k1_gemm_bu(const float* __restrict__ x,
                           const float* __restrict__ Bw,
                           float* __restrict__ ws) {
  __shared__ float Xs[2][32 * 32];       // time x k-chunk
  __shared__ float Bs[2][NSTATE * 32];   // state x k-chunk
  float* Bu = ws + BU_OFF;

  const int bt0  = blockIdx.x * 32;
  const int tid  = threadIdx.x;
  const int lane = tid & 31;
  const int wave = tid >> 5;
  const int NK   = HID / 32;             // 4

  if (wave == 0) {
    tdm_load_2d_f32((unsigned int)(size_t)Xs[0],
                    (unsigned long long)(size_t)(x + (size_t)bt0 * HID),
                    HID, 32, 32, 32, HID);
    tdm_load_2d_f32((unsigned int)(size_t)Bs[0],
                    (unsigned long long)(size_t)Bw,
                    HID, NSTATE, 32, NSTATE, HID);
  }

  v8f acc[2][2] = {{{}, {}}, {{}, {}}};  // [m-tile][state-tile]

  for (int kc = 0; kc < NK; ++kc) {
    const int cur = kc & 1;
    if (wave == 0) {
      if (kc + 1 < NK) {
        // prefetch next chunk into the alternate buffer, then wait
        // only for the current chunk (TDM completes in order: cnt<=2)
        tdm_load_2d_f32((unsigned int)(size_t)Xs[cur ^ 1],
                        (unsigned long long)(size_t)(x + (size_t)bt0 * HID + (kc + 1) * 32),
                        HID, 32, 32, 32, HID);
        tdm_load_2d_f32((unsigned int)(size_t)Bs[cur ^ 1],
                        (unsigned long long)(size_t)(Bw + (kc + 1) * 32),
                        HID, NSTATE, 32, NSTATE, HID);
        __builtin_amdgcn_s_wait_tensorcnt((short)2);
      } else {
        __builtin_amdgcn_s_wait_tensorcnt((short)0);
      }
    }
    __syncthreads();                     // current buffer ready for all waves

    v16h a0 = build_afrag_f32(Xs[cur], lane, 0);
    v16h a1 = build_afrag_f32(Xs[cur], lane, 16);
    #pragma unroll
    for (int s = 0; s < 2; ++s) {
      v16h bfrag = build_bfrag_f32(Bs[cur], lane, (wave + s * 8) * 16);
      acc[0][s] = __builtin_amdgcn_wmma_f32_16x16x32_f16(
          false, a0, false, bfrag, (short)0, acc[0][s], false, false);
      acc[1][s] = __builtin_amdgcn_wmma_f32_16x16x32_f16(
          false, a1, false, bfrag, (short)0, acc[1][s], false, false);
    }
    __syncthreads();                     // all reads of current buffer done
  }

  int Mb = (lane >= 16) ? 8 : 0;
  int N  = lane & 15;
  #pragma unroll
  for (int m = 0; m < 2; ++m) {
    #pragma unroll
    for (int s = 0; s < 2; ++s) {
      int n0 = (wave + s * 8) * 16;
      #pragma unroll
      for (int r = 0; r < 8; ++r)
        Bu[(size_t)(bt0 + m * 16 + Mb + r) * NSTATE + n0 + N] = acc[m][s][r];
    }
  }
}

// ============================================================
// k2: local scan per (batch, chunk); thread = state
// ============================================================
__global__ void k2_chunk_scan(float* __restrict__ ws) {
  const float* P = ws;
  float* Bu    = ws + BU_OFF;
  float* carry = ws + CARRY_OFF;
  int n  = threadIdx.x;
  int b  = blockIdx.x / NCHUNK;
  int ch = blockIdx.x % NCHUNK;
  float M11 = P[P_M11 + n], M12 = P[P_M12 + n];
  float M21 = P[P_M21 + n], M22 = P[P_M22 + n];
  float c1  = P[P_C1 + n],  c2  = P[P_C2 + n];
  float s1 = 0.f, s2 = 0.f;
  size_t base = ((size_t)b * SEQ + (size_t)ch * CHUNK) * NSTATE + n;
  for (int t = 0; t < CHUNK; ++t) {
    if (t + 8 < CHUNK)
      __builtin_prefetch(&Bu[base + (size_t)(t + 8) * NSTATE], 0, 1);
    float bu = Bu[base + (size_t)t * NSTATE];
    float n1 = fmaf(M11, s1, fmaf(M12, s2, c1 * bu));
    float n2 = fmaf(M21, s1, fmaf(M22, s2, c2 * bu));
    s1 = n1; s2 = n2;
  }
  size_t ci = (((size_t)b * NCHUNK + ch) * NSTATE + n) * 2;
  carry[ci + 0] = s1;
  carry[ci + 1] = s2;
}

// ============================================================
// k3: exclusive inter-chunk scan using M^CHUNK
// ============================================================
__global__ void k3_carry_scan(float* __restrict__ ws) {
  const float* P = ws;
  float* carry = ws + CARRY_OFF;
  int n = threadIdx.x;
  int b = blockIdx.x;
  float a = P[P_MP11 + n], bb = P[P_MP12 + n];
  float c = P[P_MP21 + n], d  = P[P_MP22 + n];
  float s1 = 0.f, s2 = 0.f;
  for (int ch = 0; ch < NCHUNK; ++ch) {
    size_t ci = (((size_t)b * NCHUNK + ch) * NSTATE + n) * 2;
    float e1 = carry[ci + 0], e2 = carry[ci + 1];
    carry[ci + 0] = s1;
    carry[ci + 1] = s2;
    float n1 = fmaf(a, s1, fmaf(bb, s2, e1));
    float n2 = fmaf(c, s1, fmaf(d,  s2, e2));
    s1 = n1; s2 = n2;
  }
}

// ============================================================
// k4: replay local scan with carry; write z over Bu in place
// ============================================================
__global__ void k4_apply(float* __restrict__ ws) {
  const float* P = ws;
  float* Bu    = ws + BU_OFF;
  float* carry = ws + CARRY_OFF;
  int n  = threadIdx.x;
  int b  = blockIdx.x / NCHUNK;
  int ch = blockIdx.x % NCHUNK;
  float M11 = P[P_M11 + n], M12 = P[P_M12 + n];
  float M21 = P[P_M21 + n], M22 = P[P_M22 + n];
  float c1  = P[P_C1 + n],  c2  = P[P_C2 + n];
  size_t ci = (((size_t)b * NCHUNK + ch) * NSTATE + n) * 2;
  float s1 = carry[ci + 0], s2 = carry[ci + 1];
  size_t base = ((size_t)b * SEQ + (size_t)ch * CHUNK) * NSTATE + n;
  for (int t = 0; t < CHUNK; ++t) {
    if (t + 8 < CHUNK)
      __builtin_prefetch(&Bu[base + (size_t)(t + 8) * NSTATE], 0, 1);
    float bu = Bu[base + (size_t)t * NSTATE];
    float n1 = fmaf(M11, s1, fmaf(M12, s2, c1 * bu));
    float n2 = fmaf(M21, s1, fmaf(M22, s2, c2 * bu));
    s1 = n1; s2 = n2;
    Bu[base + (size_t)t * NSTATE] = s2;
  }
}

// ============================================================
// k5: out[b,t,h] = sum_n z[b,t,n]*C[h,n] + D[h]*x[b,t,h]
// 32 time rows/block, 256 thr (8 waves); wave -> h tile, 2 m tiles;
// TDM double-buffered staging of z tile and C chunk.
// ============================================================
__global__ void k5_gemm_out(const float* __restrict__ x,
                            const float* __restrict__ Cw,
                            const float* __restrict__ Dw,
                            const float* __restrict__ ws,
                            float* __restrict__ out) {
  __shared__ float Zs[2][32 * 32];       // time x n-chunk
  __shared__ float Cs[2][HID * 32];      // h x n-chunk (Cs[h][k])
  const float* Bu = ws + BU_OFF;         // holds z now

  const int bt0  = blockIdx.x * 32;
  const int tid  = threadIdx.x;
  const int lane = tid & 31;
  const int wave = tid >> 5;
  const int h0   = wave * 16;
  const int NK   = NSTATE / 32;          // 8

  if (wave == 0) {
    tdm_load_2d_f32((unsigned int)(size_t)Zs[0],
                    (unsigned long long)(size_t)(Bu + (size_t)bt0 * NSTATE),
                    NSTATE, 32, 32, 32, NSTATE);
    tdm_load_2d_f32((unsigned int)(size_t)Cs[0],
                    (unsigned long long)(size_t)Cw,
                    NSTATE, HID, 32, HID, NSTATE);
  }

  v8f acc[2] = {{}, {}};
  for (int kc = 0; kc < NK; ++kc) {
    const int cur = kc & 1;
    if (wave == 0) {
      if (kc + 1 < NK) {
        tdm_load_2d_f32((unsigned int)(size_t)Zs[cur ^ 1],
                        (unsigned long long)(size_t)(Bu + (size_t)bt0 * NSTATE + (kc + 1) * 32),
                        NSTATE, 32, 32, 32, NSTATE);
        tdm_load_2d_f32((unsigned int)(size_t)Cs[cur ^ 1],
                        (unsigned long long)(size_t)(Cw + (kc + 1) * 32),
                        NSTATE, HID, 32, HID, NSTATE);
        __builtin_amdgcn_s_wait_tensorcnt((short)2);
      } else {
        __builtin_amdgcn_s_wait_tensorcnt((short)0);
      }
    }
    __syncthreads();

    v16h a0 = build_afrag_f32(Zs[cur], lane, 0);
    v16h a1 = build_afrag_f32(Zs[cur], lane, 16);
    v16h bfrag = build_bfrag_f32(Cs[cur], lane, h0);
    acc[0] = __builtin_amdgcn_wmma_f32_16x16x32_f16(
        false, a0, false, bfrag, (short)0, acc[0], false, false);
    acc[1] = __builtin_amdgcn_wmma_f32_16x16x32_f16(
        false, a1, false, bfrag, (short)0, acc[1], false, false);
    __syncthreads();
  }

  int Mb = (lane >= 16) ? 8 : 0;
  int N  = lane & 15;
  float Dh = Dw[h0 + N];
  #pragma unroll
  for (int m = 0; m < 2; ++m) {
    #pragma unroll
    for (int r = 0; r < 8; ++r) {
      size_t idx = (size_t)(bt0 + m * 16 + Mb + r) * HID + h0 + N;
      out[idx] = acc[m][r] + Dh * x[idx];
    }
  }
}

// ============================================================
extern "C" void kernel_launch(void* const* d_in, const int* in_sizes, int n_in,
                              void* d_out, int out_size, void* d_ws, size_t ws_size,
                              hipStream_t stream) {
  const float* x      = (const float*)d_in[0];
  const float* A_diag = (const float*)d_in[1];
  const float* steps  = (const float*)d_in[2];
  const float* Bw     = (const float*)d_in[3];
  const float* Cw     = (const float*)d_in[4];
  const float* Dw     = (const float*)d_in[5];
  float* ws  = (float*)d_ws;
  float* out = (float*)d_out;
  (void)in_sizes; (void)n_in; (void)out_size; (void)ws_size;

  k0_params    <<<1,                      NSTATE, 0, stream>>>(A_diag, steps, ws);
  k1_gemm_bu   <<<dim3(BATCH * SEQ / 32), 256,    0, stream>>>(x, Bw, ws);
  k2_chunk_scan<<<dim3(BATCH * NCHUNK),   NSTATE, 0, stream>>>(ws);
  k3_carry_scan<<<dim3(BATCH),            NSTATE, 0, stream>>>(ws);
  k4_apply     <<<dim3(BATCH * NCHUNK),   NSTATE, 0, stream>>>(ws);
  k5_gemm_out  <<<dim3(BATCH * SEQ / 32), 256,    0, stream>>>(x, Cw, Dw, ws, out);
}
